// GPTJBlock_24326694764799
// MI455X (gfx1250) — matrix operand order repaired and checked by reading, not compiled, not run on hardware
//
#include <hip/hip_runtime.h>
#include <hip/hip_bf16.h>

// ---------------------------------------------------------------------------
// GPT-J block for MI455X (gfx1250, wave32, WMMA, async-to-LDS).
// T=2048, DM=4096, H=16, HD=256, RD=64, DFF=16384
// ---------------------------------------------------------------------------

typedef __bf16 bf16;
typedef __attribute__((ext_vector_type(16))) __bf16 v16bf;
typedef __attribute__((ext_vector_type(8)))  __bf16 v8bf;
typedef __attribute__((ext_vector_type(8)))  float  v8f;
typedef __attribute__((ext_vector_type(4)))  float  v4f;
typedef __attribute__((ext_vector_type(4)))  int    v4i;

union Frag { v16bf v; v8bf h[2]; };
struct __align__(8) P4 { bf16 x[4]; };

#define AS1 __attribute__((address_space(1)))
#define AS3 __attribute__((address_space(3)))

#if __has_builtin(__builtin_amdgcn_global_load_async_to_lds_b128)
#define ASYNC_LDS 1
#else
#define ASYNC_LDS 0
#endif

// 16-byte global->LDS copy: CDNA5 async DMA path (ASYNCcnt) when available.
__device__ __forceinline__ void async_cp16(void* lds_dst, const void* g_src) {
#if ASYNC_LDS
  __builtin_amdgcn_global_load_async_to_lds_b128(
      (AS1 v4i*)g_src, (AS3 v4i*)lds_dst, 0, 0);
#else
  *(v8bf*)lds_dst = *(const v8bf*)g_src;
#endif
}

__device__ __forceinline__ void wait_async_zero() {
#if ASYNC_LDS
#if __has_builtin(__builtin_amdgcn_s_wait_asynccnt)
  __builtin_amdgcn_s_wait_asynccnt(0);
#else
  asm volatile("s_wait_asynccnt 0x0" ::: "memory");
#endif
#endif
}

__device__ __forceinline__ v8f wmma_bf16(const v16bf& a, const v16bf& b, const v8f& c) {
  // D = A(16x32 bf16) * B(32x16 bf16) + C(16x16 f32)
  return __builtin_amdgcn_wmma_f32_16x16x32_bf16(false, a, false, b, (short)0, c, false, false);
}

// Branch-free tanh-GELU: tanh(u) = 1 - 2/(1+e^{2u}) (exact at +-inf limits).
__device__ __forceinline__ float gelu_tanh(float x) {
  float u = 0.7978845608028654f * (x + 0.044715f * x * x * x);
  float e = __expf(2.f * u);
  float t = 1.f - 2.f * __builtin_amdgcn_rcpf(e + 1.f);
  return 0.5f * x * (1.f + t);
}

#define T_SEQ 2048
#define DM    4096
#define NH    16
#define HD    256
#define DFF   16384

// ---------------------------------------------------------------------------
// 1) LayerNorm -> bf16 activations (one block per row, 256 threads)
// ---------------------------------------------------------------------------
__global__ __launch_bounds__(256) void ln_kernel(
    const float* __restrict__ x, const float* __restrict__ g,
    const float* __restrict__ b, bf16* __restrict__ out)
{
  int t = blockIdx.x;
  const float* row = x + (size_t)t * DM;
  float s = 0.f, s2 = 0.f;
  for (int i = threadIdx.x; i < DM; i += 256) { float v = row[i]; s += v; s2 += v * v; }
  __shared__ float red0[256], red1[256];
  red0[threadIdx.x] = s; red1[threadIdx.x] = s2;
  __syncthreads();
  for (int off = 128; off > 0; off >>= 1) {
    if ((int)threadIdx.x < off) { red0[threadIdx.x] += red0[threadIdx.x + off];
                                  red1[threadIdx.x] += red1[threadIdx.x + off]; }
    __syncthreads();
  }
  float mu   = red0[0] * (1.f / DM);
  float var  = red1[0] * (1.f / DM) - mu * mu;
  float rstd = rsqrtf(var + 1e-5f);
  for (int i = threadIdx.x; i < DM; i += 256)
    out[(size_t)t * DM + i] = (bf16)((row[i] - mu) * rstd * g[i] + b[i]);
}

// ---------------------------------------------------------------------------
// 2) WMMA GEMM:  C[MxN](f32) = A[MxK](bf16) * B[KxN](f32->bf16)
//    128x128 wg tile, 8 waves (2Mx4N), wave 64x32, K-step 32.
//    Double-buffered LDS; A staged via async-to-LDS DMA; B prefetched to
//    registers and converted/transposed into LDS behind the WMMAs.
//    Template epilogue -> straight-line code (no per-element branches).
// ---------------------------------------------------------------------------
#define BM 128
#define BN 128
#define BK 32
#define LDA 40   // bf16 elems per A-tile row (32 + 8 pad) -> 80B, 16B aligned
#define LDB 40   // bf16 elems per B^T-tile row

template<bool BIAS, bool GELU, bool ADD, bool OUTB>
__global__ __launch_bounds__(256) void gemm_kernel(
    const bf16*  __restrict__ A, const float* __restrict__ B,
    int M, int N, int K,
    const float* __restrict__ bias,
    const float* __restrict__ add,
    float* __restrict__ outF, bf16* __restrict__ outB)
{
  __shared__ __align__(16) bf16 As[2][BM * LDA];  // [m][k] row-major
  __shared__ __align__(16) bf16 Bs[2][BN * LDB];  // [n][k] (transposed)

  const int tid  = threadIdx.x;
  const int wid  = tid >> 5;
  const int lane = tid & 31;
  const int g    = lane >> 4;     // half-wave group
  const int ln16 = lane & 15;
  const int wm   = wid >> 2;      // 0..1
  const int wn   = wid & 3;       // 0..3
  const int m0   = blockIdx.y * BM;
  const int n0   = blockIdx.x * BN;

  v8f acc[4][2] = {};

  // staging coords
  const int arow = tid >> 1, ahalf = tid & 1;          // A: 128 rows x 2 halves
  const int bnq  = (tid & 31) * 4;                     // B micro-tile col base
  const int bkq  = (tid >> 5) * 4;                     // B micro-tile row base
  const bf16* Abase = A + (size_t)(m0 + arow) * K + ahalf * 16;

  auto stageA = [&](int k0, int buf) {
    const bf16* Ap = Abase + k0;
    async_cp16(&As[buf][arow * LDA + ahalf * 16],     Ap);
    async_cp16(&As[buf][arow * LDA + ahalf * 16 + 8], Ap + 8);
  };
  auto loadB = [&](int k0, v4f r[4]) {
    #pragma unroll
    for (int i = 0; i < 4; ++i)
      r[i] = *(const v4f*)(B + (size_t)(k0 + bkq + i) * N + (n0 + bnq));
  };
  auto storeB = [&](int buf, v4f r[4]) {
    #pragma unroll
    for (int j = 0; j < 4; ++j) {
      P4 p;
      #pragma unroll
      for (int i = 0; i < 4; ++i) p.x[i] = (bf16)r[i][j];
      *(P4*)&Bs[buf][(bnq + j) * LDB + bkq] = p;
    }
  };

  const int nk = K / BK;
  v4f breg[4];

  // prologue: stage tile 0
  stageA(0, 0);
  loadB(0, breg);
  storeB(0, breg);
  wait_async_zero();
  __syncthreads();

  for (int it = 0; it < nk; ++it) {
    const int  cur  = it & 1, nxt = cur ^ 1;
    const bool more = (it + 1) < nk;
    if (more) {                       // prefetch next tile behind the WMMAs
      stageA((it + 1) * BK, nxt);     // async DMA -> LDS (ASYNCcnt)
      loadB((it + 1) * BK, breg);     // fp32 weights -> registers
    }

    // ---- fragments (CDNA5 16-bit A 16x32 / B 32x16 lane layouts)
    Frag af[4], bfr[2];
    #pragma unroll
    for (int mt = 0; mt < 4; ++mt) {
      const bf16* p = &As[cur][(wm * 64 + mt * 16 + ln16) * LDA];
      af[mt].h[0] = *(const v8bf*)(p + g * 8);        // K = g*8 .. g*8+7
      af[mt].h[1] = *(const v8bf*)(p + 16 + g * 8);   // K = 16+g*8 ..
    }
    #pragma unroll
    for (int nt = 0; nt < 2; ++nt) {
      const bf16* p = &Bs[cur][(wn * 32 + nt * 16 + ln16) * LDB];
      bfr[nt].h[0] = *(const v8bf*)(p + g * 8);       // K = g*8 .. g*8+7
      bfr[nt].h[1] = *(const v8bf*)(p + 16 + g * 8);
    }
    #pragma unroll
    for (int mt = 0; mt < 4; ++mt)
      #pragma unroll
      for (int nt = 0; nt < 2; ++nt)
        acc[mt][nt] = wmma_bf16(af[mt].v, bfr[nt].v, acc[mt][nt]);

    if (more) {
      storeB(nxt, breg);              // convert/transpose into other buffer
      wait_async_zero();              // A DMA complete before barrier
    }
    __syncthreads();
  }

  // ---- epilogue (C layout: VGPR r -> row = r + 8*g, col = lane%16)
  #pragma unroll
  for (int mt = 0; mt < 4; ++mt) {
    #pragma unroll
    for (int nt = 0; nt < 2; ++nt) {
      const int col = n0 + wn * 32 + nt * 16 + ln16;
      const float bv = BIAS ? bias[col] : 0.f;
      #pragma unroll
      for (int r = 0; r < 8; ++r) {
        const int row = m0 + wm * 64 + mt * 16 + g * 8 + r;
        float x = acc[mt][nt][r] + bv;
        if (GELU) x = gelu_tanh(x);
        const size_t idx = (size_t)row * N + col;
        if (ADD)  x += add[idx];
        if (OUTB) outB[idx] = (bf16)x;
        else      outF[idx] = x;
      }
    }
  }
}

// ---------------------------------------------------------------------------
// 3) RoPE (GPT-J interleaved, RD=64) + split QKV -> bf16 [H][T][HD]
// ---------------------------------------------------------------------------
__global__ __launch_bounds__(256) void rope_kernel(
    const float* __restrict__ qkv, const int* __restrict__ pos,
    bf16* __restrict__ Qb, bf16* __restrict__ Kb, bf16* __restrict__ Vb)
{
  const int t = blockIdx.x, h = blockIdx.y, hd = threadIdx.x;  // hd: 0..255
  const size_t src = (size_t)t * (3 * DM) + (size_t)h * HD;
  const size_t dst = ((size_t)h * T_SEQ + t) * HD + hd;
  const float p = (float)pos[t];
  float q, k;
  if (hd < 64) {
    const int i = hd >> 1;
    const float inv = __powf(10000.f, -(float)(2 * i) / 64.f);
    float s, c; __sincosf(p * inv, &s, &c);
    const float q1 = qkv[src + (hd & ~1)],      q2 = qkv[src + (hd | 1)];
    const float k1 = qkv[src + DM + (hd & ~1)], k2 = qkv[src + DM + (hd | 1)];
    if ((hd & 1) == 0) { q = q1 * c - q2 * s; k = k1 * c - k2 * s; }
    else               { q = q2 * c + q1 * s; k = k2 * c + k1 * s; }
  } else {
    q = qkv[src + hd]; k = qkv[src + DM + hd];
  }
  Qb[dst] = (bf16)q;
  Kb[dst] = (bf16)k;
  Vb[dst] = (bf16)qkv[src + 2 * DM + hd];
}

// ---------------------------------------------------------------------------
// 4) Flash-style causal attention. One workgroup = one head x 128 q rows.
//    Wave = 16 q rows; Q fragments resident; K/V tiles (32 rows) in LDS
//    (K staged via async DMA). QK^T: 16 WMMA/tile; PV: 16 WMMA/tile.
// ---------------------------------------------------------------------------
#define ALDK 264   // Ks row stride (256 + 8)
#define ALDV 40    // Vt row stride (32 + 8)
#define ALDP 40    // Ps row stride

__global__ __launch_bounds__(256) void attn_kernel(
    const bf16* __restrict__ Qb, const bf16* __restrict__ Kb,
    const bf16* __restrict__ Vb, bf16* __restrict__ ctx)
{
  __shared__ __align__(16) bf16 Ks[32 * ALDK];       // [kv][hd]
  __shared__ __align__(16) bf16 Vt[HD * ALDV];       // [hd][kv]
  __shared__ __align__(16) bf16 Ps[8 * 16 * ALDP];   // per-wave [16][kv]

  const int tid = threadIdx.x, wid = tid >> 5, lane = tid & 31;
  const int g = lane >> 4, ln16 = lane & 15;
  const int h = blockIdx.y;
  const int qbase = blockIdx.x * 128;
  const int qb = qbase + wid * 16;

  const bf16* Qh = Qb + (size_t)h * T_SEQ * HD;
  const bf16* Kh = Kb + (size_t)h * T_SEQ * HD;
  const bf16* Vh = Vb + (size_t)h * T_SEQ * HD;

  // resident Q fragments: 8 chunks of K(hd)=32
  Frag qf[8];
  {
    const bf16* qp = Qh + (size_t)(qb + ln16) * HD;
    #pragma unroll
    for (int c = 0; c < 8; ++c) {
      qf[c].h[0] = *(const v8bf*)(qp + c * 32 + g * 8);
      qf[c].h[1] = *(const v8bf*)(qp + c * 32 + 16 + g * 8);
    }
  }

  v8f O[16] = {};
  float mrow[8], lrow[8];
  #pragma unroll
  for (int r = 0; r < 8; ++r) { mrow[r] = -3.0e38f; lrow[r] = 0.f; }
  const float scale = 0.0625f;  // HD^-0.5

  const int kvend = qbase + 128;
  for (int kv0 = 0; kv0 < kvend; kv0 += 32) {
    // ---- stage K tile [32][256] row-major via async DMA
    {
      const int krow = tid >> 3, seg = tid & 7;
      const bf16* kp = Kh + (size_t)(kv0 + krow) * HD + seg * 32;
      bf16* ks = &Ks[krow * ALDK + seg * 32];
      #pragma unroll
      for (int i = 0; i < 4; ++i)
        async_cp16(ks + i * 8, kp + i * 8);
    }
    // ---- stage V transposed [hd][kv] (4kv x 8hd micro tile per thread)
    {
      const int kvb = (tid >> 5) * 4, hdb = (tid & 31) * 8;
      v8bf rows[4];
      #pragma unroll
      for (int i = 0; i < 4; ++i)
        rows[i] = *(const v8bf*)(Vh + (size_t)(kv0 + kvb + i) * HD + hdb);
      #pragma unroll
      for (int j = 0; j < 8; ++j) {
        P4 p;
        #pragma unroll
        for (int i = 0; i < 4; ++i) p.x[i] = rows[i][j];
        *(P4*)&Vt[(hdb + j) * ALDV + kvb] = p;
      }
    }
    wait_async_zero();
    __syncthreads();

    if (kv0 <= qb + 15) {   // wave-uniform causal skip
      // ---- S = Q K^T  (2 x 16x16 tiles, 8 WMMA each over hd)
      v8f S[2] = {};
      #pragma unroll
      for (int sub = 0; sub < 2; ++sub) {
        const bf16* bp = &Ks[(sub * 16 + ln16) * ALDK];
        #pragma unroll
        for (int c = 0; c < 8; ++c) {
          Frag bfr;
          bfr.h[0] = *(const v8bf*)(bp + c * 32 + g * 8);
          bfr.h[1] = *(const v8bf*)(bp + c * 32 + 16 + g * 8);
          S[sub] = wmma_bf16(qf[c].v, bfr.v, S[sub]);
        }
      }
      // ---- scale + causal mask + online softmax (16-lane shuffles)
      float sv[2][8], mnew[8], corr[8];
      #pragma unroll
      for (int sub = 0; sub < 2; ++sub) {
        const int col = kv0 + sub * 16 + ln16;
        #pragma unroll
        for (int r = 0; r < 8; ++r) {
          const int row = qb + g * 8 + r;
          float x = S[sub][r] * scale;
          if (col > row) x = -3.0e38f;
          sv[sub][r] = x;
        }
      }
      #pragma unroll
      for (int r = 0; r < 8; ++r) {
        float m = fmaxf(sv[0][r], sv[1][r]);
        for (int off = 1; off < 16; off <<= 1) m = fmaxf(m, __shfl_xor(m, off, 32));
        mnew[r] = fmaxf(mrow[r], m);
        float p0 = __expf(sv[0][r] - mnew[r]);
        float p1 = __expf(sv[1][r] - mnew[r]);
        sv[0][r] = p0; sv[1][r] = p1;
        float s = p0 + p1;
        for (int off = 1; off < 16; off <<= 1) s += __shfl_xor(s, off, 32);
        corr[r] = __expf(mrow[r] - mnew[r]);
        lrow[r] = lrow[r] * corr[r] + s;
        mrow[r] = mnew[r];
      }
      #pragma unroll
      for (int ot = 0; ot < 16; ++ot)
        #pragma unroll
        for (int r = 0; r < 8; ++r) O[ot][r] *= corr[r];

      // ---- P -> per-wave LDS, reload in A-fragment layout
      bf16* pw = &Ps[wid * 16 * ALDP];
      #pragma unroll
      for (int sub = 0; sub < 2; ++sub)
        #pragma unroll
        for (int r = 0; r < 8; ++r)
          pw[(g * 8 + r) * ALDP + sub * 16 + ln16] = (bf16)sv[sub][r];
      Frag pf;
      {
        const bf16* pr = pw + ln16 * ALDP;
        pf.h[0] = *(const v8bf*)(pr + g * 8);
        pf.h[1] = *(const v8bf*)(pr + 16 + g * 8);
      }
      // ---- O += P V  (16 WMMA across hd sub-tiles)
      #pragma unroll
      for (int ot = 0; ot < 16; ++ot) {
        const bf16* vp = &Vt[(ot * 16 + ln16) * ALDV];
        Frag vf;
        vf.h[0] = *(const v8bf*)(vp + g * 8);
        vf.h[1] = *(const v8bf*)(vp + 16 + g * 8);
        O[ot] = wmma_bf16(pf.v, vf.v, O[ot]);
      }
    }
    __syncthreads();
  }

  // ---- normalize + write ctx bf16 [t][h*HD + hd]
  #pragma unroll
  for (int r = 0; r < 8; ++r) {
    const float rl = __builtin_amdgcn_rcpf(lrow[r]);
    #pragma unroll
    for (int ot = 0; ot < 16; ++ot) {
      const int row = qb + g * 8 + r;
      const int col = ot * 16 + ln16;
      ctx[(size_t)row * DM + h * HD + col] = (bf16)(O[ot][r] * rl);
    }
  }
}

// ---------------------------------------------------------------------------
// Launcher: 7 ordered kernels on `stream`.
// Workspace layout (bytes):
//   lnb   @ 0          16,777,216   (bf16 ln activations)
//   qkv   @ 16777216  100,663,296   (f32 QKV)
//   Qb    @ 117440512  16,777,216
//   Kb    @ 134217728  16,777,216
//   Vb    @ 150994944  16,777,216
//   ctxb  @ 167772160  16,777,216
//   fcact @ 184549376  67,108,864   -> total 251,658,240 B
// ---------------------------------------------------------------------------
extern "C" void kernel_launch(void* const* d_in, const int* in_sizes, int n_in,
                              void* d_out, int out_size, void* d_ws, size_t ws_size,
                              hipStream_t stream) {
  const int*   positions = (const int*)  d_in[0];
  const float* hidden    = (const float*)d_in[1];
  const float* ln_g      = (const float*)d_in[2];
  const float* ln_b      = (const float*)d_in[3];
  const float* w_qkv     = (const float*)d_in[4];
  const float* w_out     = (const float*)d_in[5];
  const float* w_fc_in   = (const float*)d_in[6];
  const float* b_fc_in   = (const float*)d_in[7];
  const float* w_fc_out  = (const float*)d_in[8];
  const float* b_fc_out  = (const float*)d_in[9];
  float* out = (float*)d_out;

  char* ws = (char*)d_ws;
  bf16*  lnb   = (bf16*) (ws);
  float* qkv   = (float*)(ws + 16777216u);
  bf16*  Qb    = (bf16*) (ws + 117440512u);
  bf16*  Kb    = (bf16*) (ws + 134217728u);
  bf16*  Vb    = (bf16*) (ws + 150994944u);
  bf16*  ctxb  = (bf16*) (ws + 167772160u);
  bf16*  fcact = (bf16*) (ws + 184549376u);

  // 1) LayerNorm -> bf16
  ln_kernel<<<T_SEQ, 256, 0, stream>>>(hidden, ln_g, ln_b, lnb);

  // 2) QKV GEMM: [2048x4096]x[4096x12288] -> f32
  gemm_kernel<false,false,false,false><<<dim3(3 * DM / BN, T_SEQ / BM), 256, 0, stream>>>(
      lnb, w_qkv, T_SEQ, 3 * DM, DM, nullptr, nullptr, qkv, nullptr);

  // 3) RoPE + head split -> bf16 [H][T][HD]
  rope_kernel<<<dim3(T_SEQ, NH), 256, 0, stream>>>(qkv, positions, Qb, Kb, Vb);

  // 4) Causal flash attention -> ctx bf16 [T][DM]
  attn_kernel<<<dim3(T_SEQ / 128, NH), 256, 0, stream>>>(Qb, Kb, Vb, ctxb);

  // 5) Out-proj + residual: out = ctx @ w_out + hidden
  gemm_kernel<false,false,true,false><<<dim3(DM / BN, T_SEQ / BM), 256, 0, stream>>>(
      ctxb, w_out, T_SEQ, DM, DM, nullptr, hidden, out, nullptr);

  // 6) MLP in: gelu(ln @ w_fc_in + b_fc_in) -> bf16
  gemm_kernel<true,true,false,true><<<dim3(DFF / BN, T_SEQ / BM), 256, 0, stream>>>(
      lnb, w_fc_in, T_SEQ, DFF, DM, b_fc_in, nullptr, nullptr, fcact);

  // 7) MLP out, accumulate: out += fcact @ w_fc_out + b_fc_out
  gemm_kernel<true,false,true,false><<<dim3(DM / BN, T_SEQ / BM), 256, 0, stream>>>(
      fcact, w_fc_out, T_SEQ, DM, DFF, b_fc_out, out, out, nullptr);
}
